// CBR_73478300500659
// MI455X (gfx1250) — compile-verified, matrix operand order checked by reading, not run on hardware
//
#include <hip/hip_runtime.h>
#include <hip/hip_bf16.h>
#include <math.h>

typedef _Float16 f16_t;
typedef __attribute__((ext_vector_type(16))) _Float16 v16h;
typedef __attribute__((ext_vector_type(8)))  float    v8f;
typedef __attribute__((ext_vector_type(4)))  unsigned int u32x4;
typedef __attribute__((ext_vector_type(8)))  int          i32x8;
typedef __attribute__((ext_vector_type(4)))  int          i32x4;

#define B_  32
#define LV_ 196
#define LT_ 40
#define D_  512
#define H_  8
#define DH_ 64
#define ROWS_ (B_ * LV_)   // 6272

union Frag { v16h v; uint4 u[2]; };

// ---------------------------------------------------------------------------
// float -> f16 conversion
// ---------------------------------------------------------------------------
__global__ void cvt_f32_to_f16(const float* __restrict__ src, f16_t* __restrict__ dst, int n) {
    int i = blockIdx.x * blockDim.x + threadIdx.x;
    if (i < n) dst[i] = (f16_t)src[i];
}

// ---------------------------------------------------------------------------
// masked text mean: sent_mean[b,d] = (1/LT) * sum_l sent[b,l,d]*mask[b,l]
// ---------------------------------------------------------------------------
__global__ void sent_mean_kernel(const float* __restrict__ sent, const float* __restrict__ mask,
                                 float* __restrict__ out) {
    int b = blockIdx.x, d = threadIdx.x;
    float s = 0.f;
    for (int l = 0; l < LT_; ++l)
        s += sent[((size_t)b * LT_ + l) * D_ + d] * mask[b * LT_ + l];
    out[b * D_ + d] = s * (1.0f / (float)LT_);
}

// ---------------------------------------------------------------------------
// build conv1 input: A1[r, c] = c<512 ? bef[r,c] : sent_mean[r/196, c-512]  (f16)
// ---------------------------------------------------------------------------
__global__ void pack_a1_kernel(const float* __restrict__ bef, const float* __restrict__ smean,
                               f16_t* __restrict__ a1) {
    int i = blockIdx.x * blockDim.x + threadIdx.x;
    if (i >= ROWS_ * 2 * D_) return;
    int r = i >> 10;          // / 1024
    int c = i & 1023;
    int b = r / LV_;
    float v = (c < D_) ? bef[(size_t)r * D_ + c] : smean[b * D_ + (c - D_)];
    a1[i] = (f16_t)v;
}

// ---------------------------------------------------------------------------
// Generic WMMA GEMM:  C[M,N] = A[M,K] * W[N,K]^T + bias   (f16 in, f16 out, f32 acc)
// one wave per 16x16 output tile; wave-uniform control flow (EXEC all ones)
// ---------------------------------------------------------------------------
__global__ __launch_bounds__(256)
void wmma_gemm_f16(const f16_t* __restrict__ A, const f16_t* __restrict__ W,
                   const float* __restrict__ bias, f16_t* __restrict__ C,
                   int M, int N, int K) {
    int wave = (blockIdx.x * blockDim.x + threadIdx.x) >> 5;
    int lane = threadIdx.x & 31;
    int tn_count = N >> 4;
    int ntiles = (M >> 4) * tn_count;
    if (wave >= ntiles) return;                 // whole-wave uniform
    int tm = wave / tn_count, tn = wave % tn_count;
    int hh = lane >> 4, l16 = lane & 15;
    int m = tm * 16 + l16;
    int n = tn * 16 + l16;
    const f16_t* arow = A + (size_t)m * K + 8 * hh;
    const f16_t* brow = W + (size_t)n * K + 8 * hh;
    v8f acc = {};
    for (int k0 = 0; k0 < K; k0 += 32) {
        Frag a, b;
        a.u[0] = *(const uint4*)(arow + k0);
        a.u[1] = *(const uint4*)(arow + k0 + 16);
        b.u[0] = *(const uint4*)(brow + k0);
        b.u[1] = *(const uint4*)(brow + k0 + 16);
        acc = __builtin_amdgcn_wmma_f32_16x16x32_f16(false, a.v, false, b.v,
                                                     (short)0, acc, false, false);
    }
    float bv = bias ? bias[n] : 0.0f;
#pragma unroll
    for (int r = 0; r < 8; ++r) {
        int row = tm * 16 + 8 * hh + r;
        C[(size_t)row * N + n] = (f16_t)(acc[r] + bv);
    }
}

// ---------------------------------------------------------------------------
// Per-(b,h) self-attention over 196 positions. qkv f16 (row = b*196+l, 1536 wide),
// K/V staged in LDS (2*196*64 f32 = 100KB, fine on 320KB/WGP CDNA5).
// ---------------------------------------------------------------------------
__global__ void attention_kernel(const f16_t* __restrict__ qkv, f16_t* __restrict__ o) {
    extern __shared__ float smem_att[];
    float* ksh = smem_att;              // 196*64
    float* vsh = smem_att + LV_ * DH_;  // 196*64
    int b = blockIdx.x / H_, hh = blockIdx.x % H_;
    int tid = threadIdx.x;
    for (int i = tid; i < LV_ * DH_; i += 256) {
        int m = i / DH_, d = i % DH_;
        size_t base = ((size_t)(b * LV_ + m)) * (3 * D_) + hh * DH_;
        ksh[i] = (float)qkv[base + D_ + d];
        vsh[i] = (float)qkv[base + 2 * D_ + d];
    }
    __syncthreads();
    int t = tid;
    if (t < LV_) {
        const f16_t* qp = qkv + ((size_t)(b * LV_ + t)) * (3 * D_) + hh * DH_;
        float q[DH_];
#pragma unroll
        for (int d = 0; d < DH_; ++d) q[d] = (float)qp[d] * 0.125f; // 1/sqrt(64)
        float mx = -INFINITY;
        for (int m = 0; m < LV_; ++m) {
            float s = 0.f;
#pragma unroll
            for (int d = 0; d < DH_; ++d) s += q[d] * ksh[m * DH_ + d];
            mx = fmaxf(mx, s);
        }
        float sum = 0.f;
        float acc[DH_];
#pragma unroll
        for (int d = 0; d < DH_; ++d) acc[d] = 0.f;
        for (int m = 0; m < LV_; ++m) {
            float s = 0.f;
#pragma unroll
            for (int d = 0; d < DH_; ++d) s += q[d] * ksh[m * DH_ + d];
            float p = __expf(s - mx);
            sum += p;
#pragma unroll
            for (int d = 0; d < DH_; ++d) acc[d] += p * vsh[m * DH_ + d];
        }
        float inv = 1.0f / sum;
        f16_t* op = o + ((size_t)(b * LV_ + t)) * D_ + hh * DH_;
#pragma unroll
        for (int d = 0; d < DH_; ++d) op[d] = (f16_t)(acc[d] * inv);
    }
}

// ---------------------------------------------------------------------------
// CAS-based float max on LDS
// ---------------------------------------------------------------------------
__device__ inline void atomicMaxF(float* addr, float val) {
    unsigned int* ua = (unsigned int*)addr;
    unsigned int cur = __float_as_uint(*addr);
    while (__uint_as_float(cur) < val) {
        unsigned int old = atomicCAS(ua, cur, __float_as_uint(val));
        if (old == cur) break;
        cur = old;
    }
}

// ---------------------------------------------------------------------------
// Cross-batch logits block, fused max reductions.
// One block per (a,b): logits_ab = QL[a] (196x512) * KL[b]^T (196x512).
// Entire KL[b] staged in LDS (208x512 f16) by the Tensor Data Mover: the block
// is a single contiguous 200,704B region -> one 1-D TDM descriptor
// (25088 x 8B elements), issued by wave 0, completion via s_wait_tensorcnt.
// Dynamic LDS starts at offset 0 (no static LDS in this kernel), so
// D#.lds_addr = 0 targets `kl` directly.
// ---------------------------------------------------------------------------
#define LVP 208   // 13 tiles of 16
#define KLQW (LV_ * D_ * 2 / 8)   // 25088 8-byte elements in a KL block
__global__ __launch_bounds__(256)
void cross_logits_kernel(const f16_t* __restrict__ QL, const f16_t* __restrict__ KL,
                         float* __restrict__ t2v, float* __restrict__ v2t) {
    extern __shared__ char smem[];
    f16_t* kl     = (f16_t*)smem;                          // LVP*512 f16 (dyn LDS offset 0)
    float* rowmax = (float*)(smem + (size_t)LVP * D_ * 2); // LVP
    float* colmax = rowmax + LVP;                          // LVP
    int a = blockIdx.x >> 5, b = blockIdx.x & 31;
    int tid = threadIdx.x;

    // --- TDM: async copy KL[b] (contiguous 200704B) into LDS offset 0 ---
    if (tid < 32) {                                        // wave 0 only issues the DMA
        unsigned long long gaddr =
            (unsigned long long)(const void*)(KL + (size_t)b * LV_ * D_);
        u32x4 g0 = { 1u,                                   // count=1 valid descriptor
                     0u,                                   // lds_addr = 0
                     (unsigned int)gaddr,                  // global_addr[31:0]
                     (unsigned int)((gaddr >> 32) & 0x1FFFFFFu) | (2u << 30) }; // addr[56:32] | type=2
        i32x8 g1 = { 0x00030000,                           // data_size=8B, mask=0
                     (int)((KLQW & 0xFFFF) << 16),         // tensor_dim0[15:0]
                     (int)(((KLQW >> 16) & 0xFFFF) | (1 << 16)), // tensor_dim0[31:16] | tensor_dim1=1
                     (int)((KLQW & 0xFFFF) << 16),         // tensor_dim1 hi=0 | tile_dim0
                     1,                                    // tile_dim1=1, tile_dim2=0
                     (int)KLQW,                            // tensor_dim0_stride[31:0]
                     0, 0 };
        i32x4 z4 = { 0, 0, 0, 0 };
        i32x8 z8 = { 0, 0, 0, 0, 0, 0, 0, 0 };
        __builtin_amdgcn_tensor_load_to_lds(g0, g1, z4, z4, z8, 0);
        __builtin_amdgcn_s_wait_tensorcnt(0);
    }
    // zero the pad rows + init reductions (disjoint LDS, overlaps the DMA)
    {
        unsigned int* dst = (unsigned int*)kl;
        for (int i = tid; i < (LVP - LV_) * (D_ / 2); i += 256) dst[LV_ * (D_ / 2) + i] = 0u;
        for (int i = tid; i < LVP; i += 256) { rowmax[i] = -INFINITY; colmax[i] = -INFINITY; }
    }
    __syncthreads();

    int wave = tid >> 5, lane = tid & 31;
    int hh = lane >> 4, l16 = lane & 15;

    for (int tr = wave; tr < 13; tr += 8) {               // wave-uniform
        // load full K row of A fragments for this tile-row (16 chunks = 128 VGPRs)
        Frag afr[16];
        int mrow = tr * 16 + l16;
        int mclamp = mrow > (LV_ - 1) ? (LV_ - 1) : mrow; // clamp pad rows (masked later)
        const f16_t* arow = QL + ((size_t)a * LV_ + mclamp) * D_ + 8 * hh;
#pragma unroll
        for (int kc = 0; kc < 16; ++kc) {
            afr[kc].u[0] = *(const uint4*)(arow + kc * 32);
            afr[kc].u[1] = *(const uint4*)(arow + kc * 32 + 16);
        }
        for (int tc = 0; tc < 13; ++tc) {
            v8f acc = {};
            const f16_t* brow = kl + (size_t)(tc * 16 + l16) * D_ + 8 * hh;
#pragma unroll
            for (int kc = 0; kc < 16; ++kc) {
                Frag bf;
                bf.u[0] = *(const uint4*)(brow + kc * 32);
                bf.u[1] = *(const uint4*)(brow + kc * 32 + 16);
                acc = __builtin_amdgcn_wmma_f32_16x16x32_f16(false, afr[kc].v, false, bf.v,
                                                             (short)0, acc, false, false);
            }
            bool colvalid = (tc * 16 + l16) < LV_;
            // row-max: butterfly over the 16 lanes sharing a row
#pragma unroll
            for (int r = 0; r < 8; ++r) {
                int grow = tr * 16 + 8 * hh + r;
                float v = colvalid ? acc[r] : -INFINITY;
                v = fmaxf(v, __shfl_xor(v, 8, 16));
                v = fmaxf(v, __shfl_xor(v, 4, 16));
                v = fmaxf(v, __shfl_xor(v, 2, 16));
                v = fmaxf(v, __shfl_xor(v, 1, 16));
                if (l16 == 0 && grow < LV_) atomicMaxF(&rowmax[grow], v);
            }
            // col-max: fold this lane's 8 rows
            if (colvalid) {
                float cm = -INFINITY;
#pragma unroll
                for (int r = 0; r < 8; ++r) {
                    int grow = tr * 16 + 8 * hh + r;
                    if (grow < LV_) cm = fmaxf(cm, acc[r]);
                }
                atomicMaxF(&colmax[tc * 16 + l16], cm);
            }
        }
    }
    __syncthreads();
    if (tid == 0) {
        float s = 0.f;
        for (int i = 0; i < LV_; ++i) s += rowmax[i];
        t2v[a * B_ + b] = s * (1.0f / (float)LV_);
    }
    if (tid == 32) {
        float s = 0.f;
        for (int i = 0; i < LV_; ++i) s += colmax[i];
        v2t[a * B_ + b] = s * (1.0f / (float)LV_);
    }
}

// ---------------------------------------------------------------------------
// Final loss: S = 0.5*(t2v+v2t)*exp(ls); 0.5*(CE(S)+CE(S^T)) on diagonal
// ---------------------------------------------------------------------------
__global__ void loss_kernel(const float* __restrict__ t2v, const float* __restrict__ v2t,
                            const float* __restrict__ ls, float* __restrict__ out) {
    __shared__ float S[B_][B_];
    __shared__ float rl[B_], cl[B_];
    int tid = threadIdx.x;
    float e = expf(ls[0]);
    int i = tid >> 5, j = tid & 31;
    S[i][j] = 0.5f * (t2v[i * B_ + j] + v2t[i * B_ + j]) * e;
    __syncthreads();
    if (tid < B_) {
        float mx = -INFINITY;
        for (int c = 0; c < B_; ++c) mx = fmaxf(mx, S[tid][c]);
        float s = 0.f;
        for (int c = 0; c < B_; ++c) s += expf(S[tid][c] - mx);
        rl[tid] = S[tid][tid] - (mx + logf(s));
    } else if (tid < 2 * B_) {
        int c = tid - B_;
        float mx = -INFINITY;
        for (int r = 0; r < B_; ++r) mx = fmaxf(mx, S[r][c]);
        float s = 0.f;
        for (int r = 0; r < B_; ++r) s += expf(S[r][c] - mx);
        cl[c] = S[c][c] - (mx + logf(s));
    }
    __syncthreads();
    if (tid == 0) {
        float l1 = 0.f, l2 = 0.f;
        for (int k = 0; k < B_; ++k) { l1 += rl[k]; l2 += cl[k]; }
        out[0] = 0.5f * ((-l1 / (float)B_) + (-l2 / (float)B_));
    }
}

// ---------------------------------------------------------------------------
// host launch
// ---------------------------------------------------------------------------
static inline size_t alignup(size_t x) { return (x + 255) & ~(size_t)255; }

extern "C" void kernel_launch(void* const* d_in, const int* in_sizes, int n_in,
                              void* d_out, int out_size, void* d_ws, size_t ws_size,
                              hipStream_t stream) {
    const float* bef   = (const float*)d_in[0];
    const float* sent  = (const float*)d_in[1];
    const float* aft   = (const float*)d_in[2];
    const float* masks = (const float*)d_in[3];
    const float* conv1_w = (const float*)d_in[4];
    const float* conv1_b = (const float*)d_in[5];
    const float* in_proj_w = (const float*)d_in[6];
    const float* out_proj_w = (const float*)d_in[7];
    const float* conv2_w = (const float*)d_in[8];
    const float* conv2_b = (const float*)d_in[9];
    const float* q_w = (const float*)d_in[10];
    const float* q_b = (const float*)d_in[11];
    const float* k_w = (const float*)d_in[12];
    const float* k_b = (const float*)d_in[13];
    const float* logit_scale = (const float*)d_in[14];

    char* ws = (char*)d_ws;
    size_t off = 0;
    auto carve = [&](size_t bytes) { char* p = ws + off; off = alignup(off + bytes); return p; };

    float*  smean   = (float*)carve((size_t)B_ * D_ * 4);
    f16_t*  conv1h  = (f16_t*)carve((size_t)D_ * 2 * D_ * 2);
    f16_t*  inprojh = (f16_t*)carve((size_t)3 * D_ * D_ * 2);
    f16_t*  outprojh= (f16_t*)carve((size_t)D_ * D_ * 2);
    f16_t*  conv2h  = (f16_t*)carve((size_t)D_ * D_ * 2);
    f16_t*  qwh     = (f16_t*)carve((size_t)D_ * D_ * 2);
    f16_t*  kwh     = (f16_t*)carve((size_t)D_ * D_ * 2);
    f16_t*  afth    = (f16_t*)carve((size_t)ROWS_ * D_ * 2);
    f16_t*  a1h     = (f16_t*)carve((size_t)ROWS_ * 2 * D_ * 2);
    f16_t*  v1h     = (f16_t*)carve((size_t)ROWS_ * D_ * 2);
    f16_t*  qkvh    = (f16_t*)carve((size_t)ROWS_ * 3 * D_ * 2);
    f16_t*  oh      = (f16_t*)carve((size_t)ROWS_ * D_ * 2);
    f16_t*  o2h     = (f16_t*)carve((size_t)ROWS_ * D_ * 2);
    f16_t*  modh    = (f16_t*)carve((size_t)ROWS_ * D_ * 2);
    f16_t*  qlh     = (f16_t*)carve((size_t)ROWS_ * D_ * 2);
    f16_t*  klh     = (f16_t*)carve((size_t)ROWS_ * D_ * 2);
    float*  t2v     = (float*)carve((size_t)B_ * B_ * 4);
    float*  v2t     = (float*)carve((size_t)B_ * B_ * 4);
    (void)ws_size;

    auto cvt = [&](const float* s, f16_t* d, int n) {
        cvt_f32_to_f16<<<(n + 255) / 256, 256, 0, stream>>>(s, d, n);
    };
    auto gemm = [&](const f16_t* A, const f16_t* W, const float* bias, f16_t* C,
                    int M, int N, int K) {
        int ntiles = (M / 16) * (N / 16);
        wmma_gemm_f16<<<(ntiles + 7) / 8, 256, 0, stream>>>(A, W, bias, C, M, N, K);
    };

    // 0) masked text mean
    sent_mean_kernel<<<B_, D_, 0, stream>>>(sent, masks, smean);

    // 1) f16 conversions
    cvt(conv1_w,    conv1h,   D_ * 2 * D_);
    cvt(in_proj_w,  inprojh,  3 * D_ * D_);
    cvt(out_proj_w, outprojh, D_ * D_);
    cvt(conv2_w,    conv2h,   D_ * D_);
    cvt(q_w,        qwh,      D_ * D_);
    cvt(k_w,        kwh,      D_ * D_);
    cvt(aft,        afth,     ROWS_ * D_);

    // 2) conv1 input assembly + GEMM chain
    {
        int n = ROWS_ * 2 * D_;
        pack_a1_kernel<<<(n + 255) / 256, 256, 0, stream>>>(bef, smean, a1h);
    }
    gemm(a1h, conv1h, conv1_b, v1h, ROWS_, D_, 2 * D_);       // conv1
    gemm(v1h, inprojh, nullptr, qkvh, ROWS_, 3 * D_, D_);     // in_proj

    // 3) attention
    attention_kernel<<<B_ * H_, 256, 2 * LV_ * DH_ * sizeof(float), stream>>>(qkvh, oh);

    gemm(oh,  outprojh, nullptr, o2h,  ROWS_, D_, D_);        // out_proj
    gemm(o2h, conv2h,  conv2_b,  modh, ROWS_, D_, D_);        // conv2
    gemm(modh, qwh,    q_b,      qlh,  ROWS_, D_, D_);        // q proj
    gemm(afth, kwh,    k_b,      klh,  ROWS_, D_, D_);        // k proj

    // 4) cross-batch logits with fused max reductions (213KB LDS / block, TDM-staged)
    {
        size_t smem = (size_t)LVP * D_ * 2 + 2 * LVP * sizeof(float);
        cross_logits_kernel<<<B_ * B_, 256, smem, stream>>>(qlh, klh, t2v, v2t);
    }

    // 5) final scalar loss
    loss_kernel<<<1, B_ * B_, 0, stream>>>(t2v, v2t, logit_scale, (float*)d_out);
}
